// KANLayer_22625887715846
// MI455X (gfx1250) — compile-verified, matrix operand order checked by reading, not run on hardware
//
#include <hip/hip_runtime.h>

typedef _Float16 v16h __attribute__((ext_vector_type(16)));
typedef float    v8f  __attribute__((ext_vector_type(8)));

#define IN_CH   256
#define OUT_CH  256
#define GPTS    5
#define KDIM    (IN_CH * GPTS)        // 1280 = 40 * 32
#define KCHUNKS (KDIM / 32)           // 40
#define MTILE   64                    // rows per block (4 x 16)
#define EPSF    1e-8f

#define WMMA_F16(A, B, C) \
    __builtin_amdgcn_wmma_f32_16x16x32_f16(false, (A), false, (B), (short)0, (C), false, false)

// ---------- monotonic float<->uint encoding for exact atomic min/max ----------
__device__ __forceinline__ unsigned enc_f32(float f) {
    unsigned u = __float_as_uint(f);
    return (u & 0x80000000u) ? ~u : (u | 0x80000000u);
}
__device__ __forceinline__ float dec_f32(unsigned k) {
    unsigned u = (k & 0x80000000u) ? (k ^ 0x80000000u) : ~k;
    return __uint_as_float(u);
}

// ---------- CDNA5 16-bit operand fragment layout (ISA 7.12.2) ----------------
__device__ __forceinline__ int frag_off(int row, int kk) {
    int g8   = kk >> 3;                              // 8-wide K group 0..3
    int lane = (row & 15) + ((g8 & 1) << 4);
    int v    = ((kk & 7) >> 1) + ((g8 >> 1) << 2);   // VGPR index 0..7
    return lane * 16 + v * 2 + (kk & 1);             // 16 halfs per lane
}

// ============================ kernel 1: init ws ==============================
__global__ void kan_init(unsigned* ws) {
    ws[0] = 0xFFFFFFFFu;   // min key
    ws[1] = 0x00000000u;   // max key
}

// ============================ kernel 2: min/max ==============================
__global__ void kan_minmax(const float* __restrict__ x, int n, unsigned* __restrict__ ws) {
    __shared__ float smin[256];
    __shared__ float smax[256];
    int tid = threadIdx.x;
    float lmin =  3.402823466e+38f;
    float lmax = -3.402823466e+38f;
    for (int i = blockIdx.x * blockDim.x + tid; i < n; i += gridDim.x * blockDim.x) {
        float v = x[i];
        lmin = fminf(lmin, v);
        lmax = fmaxf(lmax, v);
    }
    smin[tid] = lmin; smax[tid] = lmax;
    __syncthreads();
    for (int s = 128; s > 0; s >>= 1) {
        if (tid < s) {
            smin[tid] = fminf(smin[tid], smin[tid + s]);
            smax[tid] = fmaxf(smax[tid], smax[tid + s]);
        }
        __syncthreads();
    }
    if (tid == 0) {
        atomicMin(&ws[0], enc_f32(smin[0]));
        atomicMax(&ws[1], enc_f32(smax[0]));
    }
}

// ==================== kernel 3: pack W = coef*scale to f16 frags =============
__global__ void kan_packw(const float* __restrict__ coef,
                          const float* __restrict__ scale,
                          _Float16* __restrict__ wpack) {
    int t = blockIdx.x * blockDim.x + threadIdx.x;
    if (t >= OUT_CH * KDIM) return;
    int o = t / KDIM;          // output channel = N index
    int k = t - o * KDIM;      // = i*5 + g
    int i = k / GPTS;
    float w = coef[(size_t)o * KDIM + k] * scale[(size_t)o * IN_CH + i];
    int idx = ((o >> 4) * KCHUNKS + (k >> 5)) * 512 + frag_off(o, k & 31);
    wpack[idx] = (_Float16)w;
}

// ======================= fragment set for one K-chunk ========================
struct Frags {
    v16h a0, a1;   // two A (M) fragments from LDS
    v16h b[4];     // four B (N) fragments from global (L2-resident)
};

__device__ __forceinline__ void load_frags(Frags& f,
                                           const _Float16* a0p,
                                           const _Float16* a1p,
                                           const _Float16* bp,
                                           int kc) {
    f.a0 = *(const v16h*)(a0p + kc * 512);
    f.a1 = *(const v16h*)(a1p + kc * 512);
#pragma unroll
    for (int j = 0; j < 4; ++j)
        f.b[j] = *(const v16h*)(bp + ((size_t)j * KCHUNKS + kc) * 512);
}

__device__ __forceinline__ void mma_frags(v8f acc[2][4], const Frags& f) {
#pragma unroll
    for (int j = 0; j < 4; ++j) {
        acc[0][j] = WMMA_F16(f.a0, f.b[j], acc[0][j]);
        acc[1][j] = WMMA_F16(f.a1, f.b[j], acc[1][j]);
    }
}

// ==================== kernel 4: fused basis + WMMA GEMM ======================
// Block: 256 threads = 8 waves = 2 M-pairs x 4 N-columns.
// Each wave: 2 M-tiles x 4 N-tiles = 8 f32 accumulators; B frags reused x2.
// Dynamic LDS: 64 x 1280 f16 basis panel in A-fragment layout (160 KB).
__global__ void __launch_bounds__(256)
kan_gemm(const float* __restrict__ x,
         const float* __restrict__ gridp,
         const float* __restrict__ bias,
         const _Float16* __restrict__ wpack,
         const unsigned* __restrict__ ws,
         float* __restrict__ out) {
    extern __shared__ _Float16 apanel[];   // [4*KCHUNKS frags][512 halfs]

    const int b0 = blockIdx.x * MTILE;

    // ---- phase A: basis panel for 64 rows, written in A-frag layout ----
    const float xmin = dec_f32(ws[0]);
    const float xmax = dec_f32(ws[1]);
    const float inv2 = 2.0f / (xmax - xmin + EPSF);
    float gv[GPTS];
#pragma unroll
    for (int g = 0; g < GPTS; ++g) gv[g] = gridp[g];

    for (int p = threadIdx.x; p < MTILE * IN_CH; p += 256) {
        int m = p >> 8;            // row in tile (IN_CH == 256)
        int i = p & (IN_CH - 1);   // input channel
        float xv = x[(size_t)(b0 + m) * IN_CH + i];
        float xn = (xv - xmin) * inv2 - 1.0f;
        float bas[GPTS];
        float s = 0.0f;
#pragma unroll
        for (int g = 0; g < GPTS; ++g) {
            float d  = fabsf(xn - gv[g]);
            float bb = (d < 1.0f) ? (1.0f - d * d * d) : 0.0f;
            bas[g] = bb;
            s += bb;
        }
        float r = 1.0f / (s + EPSF);
        int kbase = i * GPTS;
#pragma unroll
        for (int g = 0; g < GPTS; ++g) {
            int k = kbase + g;
            apanel[((m >> 4) * KCHUNKS + (k >> 5)) * 512 + frag_off(m, k & 31)] =
                (_Float16)(bas[g] * r);
        }
    }
    __syncthreads();

    // ---- phase B: ping-pong pipelined WMMA GEMM over K = 40 chunks of 32 ----
    const int lane = threadIdx.x & 31;
    const int wv   = threadIdx.x >> 5;
    const int mp   = wv & 1;        // M pair 0..1 -> m-tiles {2*mp, 2*mp+1}
    const int ncol = wv >> 1;       // N column 0..3 -> n-tiles ncol*4 .. +3

    v8f acc[2][4] = {};

    const _Float16* a0p = apanel + (size_t)(mp * 2 + 0) * KCHUNKS * 512 + lane * 16;
    const _Float16* a1p = apanel + (size_t)(mp * 2 + 1) * KCHUNKS * 512 + lane * 16;
    const _Float16* bp  = wpack  + (size_t)(ncol * 4) * KCHUNKS * 512 + lane * 16;

    Frags f0, f1;
    load_frags(f0, a0p, a1p, bp, 0);

    // unroll-by-2 ping-pong: no register rotation movs, no WMMA->VALU hazards
#pragma unroll 1
    for (int kc = 0; kc + 2 < KCHUNKS; kc += 2) {
        load_frags(f1, a0p, a1p, bp, kc + 1);
        mma_frags(acc, f0);
        load_frags(f0, a0p, a1p, bp, kc + 2);
        mma_frags(acc, f1);
    }
    // tail: chunks KCHUNKS-2 (in f0) and KCHUNKS-1
    load_frags(f1, a0p, a1p, bp, KCHUNKS - 1);
    mma_frags(acc, f0);
    mma_frags(acc, f1);

    // ---- epilogue: C layout (ISA 7.12.2): VGPR r -> row r + 8*(lane>>4) ----
    const int cb = ncol * 64 + (lane & 15);
#pragma unroll
    for (int mi = 0; mi < 2; ++mi) {
        const int rbase = b0 + (mp * 2 + mi) * 16 + ((lane >> 4) << 3);
#pragma unroll
        for (int j = 0; j < 4; ++j) {
            int col = cb + j * 16;
            float bv = bias[col];
#pragma unroll
            for (int r = 0; r < 8; ++r) {
                out[(size_t)(rbase + r) * OUT_CH + col] = acc[mi][j][r] + bv;
            }
        }
    }
}

// ================================ launcher ===================================
extern "C" void kernel_launch(void* const* d_in, const int* in_sizes, int n_in,
                              void* d_out, int out_size, void* d_ws, size_t ws_size,
                              hipStream_t stream) {
    const float* x     = (const float*)d_in[0];
    const float* gridp = (const float*)d_in[1];
    const float* coef  = (const float*)d_in[2];
    const float* scale = (const float*)d_in[3];
    const float* bias  = (const float*)d_in[4];
    float* out = (float*)d_out;

    const int nx = in_sizes[0];          // B * IN_CH
    const int B  = nx / IN_CH;           // 32768

    unsigned* ws_u = (unsigned*)d_ws;
    _Float16* wpack = (_Float16*)((char*)d_ws + 256);   // 640 KB packed W

    kan_init<<<1, 1, 0, stream>>>(ws_u);
    kan_minmax<<<2048, 256, 0, stream>>>(x, nx, ws_u);
    kan_packw<<<(OUT_CH * KDIM + 255) / 256, 256, 0, stream>>>(coef, scale, wpack);

    const size_t lds_bytes = (size_t)4 * KCHUNKS * 512 * sizeof(_Float16); // 160 KB
    kan_gemm<<<B / MTILE, 256, lds_bytes, stream>>>(x, gridp, bias, wpack, ws_u, out);
}
